// HRNNtagger_76579266887804
// MI455X (gfx1250) — matrix operand order repaired and checked by reading, not compiled, not run on hardware
//
#include <hip/hip_runtime.h>

// ---------------- problem constants ----------------
#define BDIM 64
#define TDIM 512
#define HDIM 768
#define RROWS (BDIM * TDIM) // 32768 rows of (b,t)
#define XS_LD 776           // LDS row stride (768 + 8 bf16 pad -> bank-conflict-free frags)

// ---------------- CDNA5 WMMA types ----------------
typedef __attribute__((ext_vector_type(16))) __bf16 vbf16;
typedef __attribute__((ext_vector_type(8)))  __bf16 vbf8;
typedef __attribute__((ext_vector_type(4)))  __bf16 vbf4;
typedef __attribute__((ext_vector_type(8)))  float  vf8;
typedef __attribute__((ext_vector_type(4)))  float  vf4; // clang ext_vector: OK for NT builtins

__device__ __forceinline__ __bf16 to_bf16(float f) { return (__bf16)f; }

__device__ __forceinline__ vf8 wmma_bf16(vbf16 a, vbf16 b, vf8 c) {
    // D = A(16x32 bf16) * B(32x16 bf16) + C(16x16 f32); wave32
    return __builtin_amdgcn_wmma_f32_16x16x32_bf16(
        /*neg_a=*/false, a, /*neg_b=*/false, b,
        /*c_mod=*/(short)0, c, /*reuse_a=*/false, /*reuse_b=*/false);
}

// Load A fragment (16x32 tile) from bf16 row-major src (global or LDS).
// ISA layout: lane m = lane&15 is row M=m; lanes<16 hold K = k0..k0+7 & k0+16..23,
// lanes>=16 hold K = k0+8..15 & k0+24..31  -> two contiguous 16B chunks per lane.
__device__ __forceinline__ vbf16 load_a_bf16(const __bf16* base,
                                             int ld, int row0, int k0, int lane) {
    const int m = lane & 15, hx = lane >> 4;
    const __bf16* p = base + (size_t)(row0 + m) * ld + k0 + hx * 8;
    vbf8 c0 = *(const vbf8*)(p);
    vbf8 c1 = *(const vbf8*)(p + 16);
    vbf16 a;
#pragma unroll
    for (int i = 0; i < 8; ++i) { a[i] = c0[i]; a[8 + i] = c1[i]; }
    return a;
}

// Load B fragment for C = X @ W^T : B[k,n] = W[n,k], W row-major (H,H) bf16.
// lane n = lane&15 is column; lanes<16 hold K=k0..k0+7 & k0+16..23, lanes>=16 hold
// K=k0+8..15 & k0+24..31 -> two contiguous 16B chunks of W row (col0+n).
__device__ __forceinline__ vbf16 load_b_wT(const __bf16* __restrict__ W,
                                           int ld, int col0, int k0, int lane) {
    const int n = lane & 15, hx = lane >> 4;
    const __bf16* p = W + (size_t)(col0 + n) * ld + k0 + hx * 8;
    vbf8 c0 = *(const vbf8*)(p);
    vbf8 c1 = *(const vbf8*)(p + 16);
    vbf16 b;
#pragma unroll
    for (int i = 0; i < 8; ++i) { b[i] = c0[i]; b[8 + i] = c1[i]; }
    return b;
}

// ---------------- kernel 1: convert 5 weight matrices to bf16 ----------------
__global__ __launch_bounds__(256)
void wconv_kernel(const float* __restrict__ w_ih11, const float* __restrict__ w_ih12,
                  const float* __restrict__ w_hh11, const float* __restrict__ w_ih21,
                  const float* __restrict__ w_hh21,
                  __bf16* __restrict__ o_ih11, __bf16* __restrict__ o_ih12,
                  __bf16* __restrict__ o_hh11, __bf16* __restrict__ o_ih21,
                  __bf16* __restrict__ o_hh21) {
    int i = blockIdx.x * 256 + threadIdx.x;
    if (i < HDIM * HDIM) {
        o_ih11[i] = to_bf16(w_ih11[i]);
        o_ih12[i] = to_bf16(w_ih12[i]);
        o_hh11[i] = to_bf16(w_hh11[i]);
        o_ih21[i] = to_bf16(w_ih21[i]);
        o_hh21[i] = to_bf16(w_hh21[i]);
    }
}

// ---------------- kernel 2: c12[h] = b12[h] + h_init @ W_hh12^T ----------------
__global__ __launch_bounds__(256)
void vecprep_kernel(const float* __restrict__ w_hh12, const float* __restrict__ b12,
                    const float* __restrict__ h_init, float* __restrict__ c12) {
    int h = blockIdx.x * 256 + threadIdx.x;
    if (h < HDIM) {
        const float* wrow = w_hh12 + (size_t)h * HDIM;
        float s = b12[h];
        for (int e = 0; e < HDIM; ++e) s += h_init[e] * wrow[e];
        c12[h] = s;
    }
}

// ---------------- kernel 3: zero initial states (parity-0 buffers) ----------------
__global__ __launch_bounds__(256)
void zero_state_kernel(float* __restrict__ h1f, float* __restrict__ h2f,
                       __bf16* __restrict__ h1b, __bf16* __restrict__ h2b) {
    int i = blockIdx.x * 256 + threadIdx.x;
    if (i < BDIM * HDIM) {
        h1f[i] = 0.0f; h2f[i] = 0.0f;
        h1b[i] = to_bf16(0.0f); h2b[i] = to_bf16(0.0f);
    }
}

// ---------------- kernel 4: batched GEMMs over all 32768 rows ----------------
// xp11 = X @ W_ih11^T + b11 ; h12 = tanh(X @ W_ih12^T + c12)
// One WG owns 64 rows x ALL 768 cols: X slab converted to bf16 ONCE into LDS
// (read exactly once from HBM, non-temporal), then 24 column chunks of WMMA with
// A fragments served from LDS. 8 waves = 4 row-tiles x 2 col-tiles.
__global__ __launch_bounds__(256)
void precompute_kernel(const float* __restrict__ X,
                       const __bf16* __restrict__ Wb11, const __bf16* __restrict__ Wb12,
                       const float* __restrict__ b11, const float* __restrict__ c12,
                       float* __restrict__ xp11, float* __restrict__ h12) {
    extern __shared__ __bf16 xs[]; // [64][XS_LD]
    const int tid = threadIdx.x;
    const size_t rbase = (size_t)blockIdx.x * 64;

    // ---- stage: X[rbase..rbase+64) fp32 -> bf16 LDS (streaming, non-temporal)
    const float* xsrc = X + rbase * HDIM;
#pragma unroll 4
    for (int i = 0; i < (64 * HDIM) / (256 * 4); ++i) {
        int idx = (i * 256 + tid) * 4;             // 4 floats, same row (768 % 4 == 0)
        int row = idx / HDIM, col = idx % HDIM;
        vf4 v = __builtin_nontemporal_load((const vf4*)(xsrc + idx));
        vbf4 b; b[0] = to_bf16(v.x); b[1] = to_bf16(v.y);
                b[2] = to_bf16(v.z); b[3] = to_bf16(v.w);
        *(vbf4*)(xs + (size_t)row * XS_LD + col) = b;
    }
    __syncthreads();

    const int lane = tid & 31;
    const int wave = tid >> 5;
    const int row0 = (wave & 3) * 16;            // 4 row tiles cover 64 rows
    const int m = lane & 15, hx = lane >> 4;

    for (int cb = 0; cb < HDIM / 32; ++cb) {     // 24 column chunks of 32
        const int col0 = cb * 32 + (wave >> 2) * 16;
        vf8 acc1 = {}; vf8 acc2 = {};
#pragma unroll 4
        for (int k0 = 0; k0 < HDIM; k0 += 32) {
            vbf16 a  = load_a_bf16(xs, XS_LD, row0, k0, lane);  // ds_load, shared A
            vbf16 b1 = load_b_wT(Wb11, HDIM, col0, k0, lane);
            vbf16 b2 = load_b_wT(Wb12, HDIM, col0, k0, lane);
            acc1 = wmma_bf16(a, b1, acc1);
            acc2 = wmma_bf16(a, b2, acc2);
        }
        const int n = col0 + m;
        const float bias1 = b11[n];
        const float bias2 = c12[n];
#pragma unroll
        for (int r = 0; r < 8; ++r) {
            size_t idx = (rbase + row0 + r + hx * 8) * (size_t)HDIM + n;
            // written once, read once (much later): keep out of near caches
            __builtin_nontemporal_store(acc1[r] + bias1, xp11 + idx);
            __builtin_nontemporal_store(tanhf(acc2[r] + bias2), h12 + idx);
        }
    }
}

// ---------------- kernel 5: tag_x[r] = X[r,:] @ wt_x + b_tag ----------------
__global__ __launch_bounds__(256)
void tagx_kernel(const float* __restrict__ X, const float* __restrict__ W_tag,
                 const float* __restrict__ b_tag, float* __restrict__ tagx) {
    const int r = blockIdx.x * 8 + (threadIdx.x >> 5); // one wave32 per row
    const int lane = threadIdx.x & 31;
    const float* wt_x = W_tag + 2 * HDIM;
    const float* row = X + (size_t)r * HDIM;
    float s = 0.0f;
    for (int e = lane; e < HDIM; e += 32) s += row[e] * wt_x[e];
#pragma unroll
    for (int off = 16; off > 0; off >>= 1) s += __shfl_down(s, off, 32);
    if (lane == 0) tagx[r] = s + b_tag[0];
}

// ---------------- kernel 6: one recurrence step ----------------
// WG = 256 threads = 8 waves covering ALL 64 batch rows x 32 H-cols; grid.x = 24.
// Weights/states are L2-resident (<6 MB total vs 192 MB L2); the chain is
// launch-latency bound, so we only add prefetch of the t-slices.
__global__ __launch_bounds__(256)
void step_kernel(int t,
                 const float* __restrict__ xp11, const float* __restrict__ h12,
                 const float* __restrict__ tagx,
                 const __bf16* __restrict__ Whh11b, const __bf16* __restrict__ Wih21b,
                 const __bf16* __restrict__ Whh21b,
                 const float* __restrict__ W_tag, const float* __restrict__ b21,
                 const float* __restrict__ h1f_old, const float* __restrict__ h2f_old,
                 const __bf16* __restrict__ h1b_old, const __bf16* __restrict__ h2b_old,
                 float* __restrict__ h1f_new, float* __restrict__ h2f_new,
                 __bf16* __restrict__ h1b_new, __bf16* __restrict__ h2b_new,
                 float* __restrict__ out_os, float* __restrict__ out_h2,
                 float* __restrict__ out_tagrep) {
    __shared__ float red[256];
    __shared__ float o_s[BDIM];

    const int tid = threadIdx.x;
    const int lane = tid & 31;
    const int wave = tid >> 5;
    const int row0 = (wave & 3) * 16;                     // batch tile
    const int col0 = blockIdx.x * 32 + (wave >> 2) * 16;  // H tile
    const int m = lane & 15, hx = lane >> 4;
    const int n = col0 + m;

    // warm the epilogue's strided t-slice loads (global_prefetch_b8)
#pragma unroll
    for (int r = 0; r < 8; ++r) {
        size_t tidx = ((size_t)(row0 + r + hx * 8) * TDIM + t) * HDIM + n;
        __builtin_prefetch(xp11 + tidx, 0, 0);
        __builtin_prefetch(h12 + tidx, 0, 0);
    }

    // ---- phase 1: gate o[b] = sigmoid(tag_x + h1.wt_h1 + h2.wt_h2), 4 threads/row
    {
        const int b = tid >> 2, part = tid & 3;
        const float* h1row = h1f_old + (size_t)b * HDIM;
        const float* h2row = h2f_old + (size_t)b * HDIM;
        const float* wt1 = W_tag;
        const float* wt2 = W_tag + HDIM;
        float s = 0.0f;
        const int e0 = part * (HDIM / 4);
        for (int e = e0; e < e0 + HDIM / 4; ++e)
            s += h1row[e] * wt1[e] + h2row[e] * wt2[e];
        red[tid] = s;
        __syncthreads();
        if (part == 0) {
            float tag = tagx[(size_t)b * TDIM + t] +
                        red[tid] + red[tid + 1] + red[tid + 2] + red[tid + 3];
            float o = 1.0f / (1.0f + expf(-tag));
            o_s[b] = o;
            if (blockIdx.x == 0) {
                out_os[(size_t)b * TDIM + t] = o;       // output_seq
                out_tagrep[(size_t)b * TDIM + t] = tag; // tag_rep_seq
            }
        }
        __syncthreads();
    }

    // ---- phase 2: the three recurrent GEMMs via WMMA (uniform control flow)
    vf8 acc11 = {}; // h1 @ W_hh11^T
    vf8 acc21 = {}; // h1 @ W_ih21^T + h2 @ W_hh21^T
#pragma unroll 4
    for (int k0 = 0; k0 < HDIM; k0 += 32) {
        vbf16 a1 = load_a_bf16(h1b_old, HDIM, row0, k0, lane);
        vbf16 a2 = load_a_bf16(h2b_old, HDIM, row0, k0, lane);
        vbf16 w1 = load_b_wT(Whh11b, HDIM, col0, k0, lane);
        vbf16 w2 = load_b_wT(Wih21b, HDIM, col0, k0, lane);
        vbf16 w3 = load_b_wT(Whh21b, HDIM, col0, k0, lane);
        acc11 = wmma_bf16(a1, w1, acc11);
        acc21 = wmma_bf16(a1, w2, acc21);
        acc21 = wmma_bf16(a2, w3, acc21);
    }

    // ---- epilogue: gating + state update + h2 output
    const float bias21 = b21[n];
#pragma unroll
    for (int r = 0; r < 8; ++r) {
        int b = row0 + r + hx * 8;
        size_t sidx = (size_t)b * HDIM + n;
        size_t tidx = ((size_t)b * TDIM + t) * HDIM + n;
        float o = o_s[b];
        float h11 = tanhf(acc11[r] + xp11[tidx]);
        float h21 = tanhf(acc21[r] + bias21);
        float h1n = h11 * (1.0f - o) + h12[tidx] * o;
        float h2n = h2f_old[sidx] * (1.0f - o) + h21 * o;
        h1f_new[sidx] = h1n;
        h2f_new[sidx] = h2n;
        h1b_new[sidx] = to_bf16(h1n);
        h2b_new[sidx] = to_bf16(h2n);
        __builtin_nontemporal_store(h2n, out_h2 + tidx); // h2_matrix (write-only)
    }
}

// ---------------- host-side launcher ----------------
extern "C" void kernel_launch(void* const* d_in, const int* in_sizes, int n_in,
                              void* d_out, int out_size, void* d_ws, size_t ws_size,
                              hipStream_t stream) {
    (void)in_sizes; (void)n_in; (void)out_size; (void)ws_size;

    const float* bert   = (const float*)d_in[0];
    const float* W_ih11 = (const float*)d_in[1];
    const float* W_hh11 = (const float*)d_in[2];
    const float* b11    = (const float*)d_in[3];
    const float* W_ih12 = (const float*)d_in[4];
    const float* W_hh12 = (const float*)d_in[5];
    const float* b12    = (const float*)d_in[6];
    const float* W_ih21 = (const float*)d_in[7];
    const float* W_hh21 = (const float*)d_in[8];
    const float* b21    = (const float*)d_in[9];
    const float* W_tag  = (const float*)d_in[10];
    const float* b_tag  = (const float*)d_in[11];
    const float* h_init = (const float*)d_in[12];

    float* out_os     = (float*)d_out;                        // (B,T)
    float* out_h2     = out_os + (size_t)BDIM * TDIM;         // (B,T,H)
    float* out_tagrep = out_h2 + (size_t)BDIM * TDIM * HDIM;  // (B,T,1)

    // workspace carve-up (256B aligned)
    char* ws = (char*)d_ws;
    size_t off = 0;
    auto take = [&](size_t bytes) -> char* {
        char* p = ws + off;
        off = (off + bytes + 255) & ~(size_t)255;
        return p;
    };
    float*  xp11   = (float*)take((size_t)RROWS * HDIM * 4);
    float*  h12    = (float*)take((size_t)RROWS * HDIM * 4);
    float*  tagx   = (float*)take((size_t)RROWS * 4);
    __bf16* Wb11   = (__bf16*)take((size_t)HDIM * HDIM * 2);
    __bf16* Wb12   = (__bf16*)take((size_t)HDIM * HDIM * 2);
    __bf16* Whh11b = (__bf16*)take((size_t)HDIM * HDIM * 2);
    __bf16* Wih21b = (__bf16*)take((size_t)HDIM * HDIM * 2);
    __bf16* Whh21b = (__bf16*)take((size_t)HDIM * HDIM * 2);
    float*  c12    = (float*)take((size_t)HDIM * 4);
    float*  h1f[2] = { (float*)take((size_t)BDIM * HDIM * 4),
                       (float*)take((size_t)BDIM * HDIM * 4) };
    float*  h2f[2] = { (float*)take((size_t)BDIM * HDIM * 4),
                       (float*)take((size_t)BDIM * HDIM * 4) };
    __bf16* h1b[2] = { (__bf16*)take((size_t)BDIM * HDIM * 2),
                       (__bf16*)take((size_t)BDIM * HDIM * 2) };
    __bf16* h2b[2] = { (__bf16*)take((size_t)BDIM * HDIM * 2),
                       (__bf16*)take((size_t)BDIM * HDIM * 2) };

    // 1) weights -> bf16
    wconv_kernel<<<(HDIM * HDIM + 255) / 256, 256, 0, stream>>>(
        W_ih11, W_ih12, W_hh11, W_ih21, W_hh21,
        Wb11, Wb12, Whh11b, Wih21b, Whh21b);
    // 2) c12 = b12 + h_init @ W_hh12^T
    vecprep_kernel<<<(HDIM + 255) / 256, 256, 0, stream>>>(W_hh12, b12, h_init, c12);
    // 3) zero parity-0 states
    zero_state_kernel<<<(BDIM * HDIM + 255) / 256, 256, 0, stream>>>(
        h1f[0], h2f[0], h1b[0], h2b[0]);
    // 4) batched GEMMs: xp11, h12 (64-row slabs, X staged once through LDS)
    precompute_kernel<<<RROWS / 64, 256, 64 * XS_LD * sizeof(__bf16), stream>>>(
        bert, Wb11, Wb12, b11, c12, xp11, h12);
    // 5) tag_x
    tagx_kernel<<<RROWS / 8, 256, 0, stream>>>(bert, W_tag, b_tag, tagx);
    // 6) the 512-step recurrence (ping-pong state buffers)
    for (int t = 0; t < TDIM; ++t) {
        int p = t & 1, q = p ^ 1;
        step_kernel<<<HDIM / 32, 256, 0, stream>>>(
            t, xp11, h12, tagx, Whh11b, Wih21b, Whh21b, W_tag, b21,
            h1f[p], h2f[p], h1b[p], h2b[p],
            h1f[q], h2f[q], h1b[q], h2b[q],
            out_os, out_h2, out_tagrep);
    }
}
